// Attention_83167746719747
// MI455X (gfx1250) — compile-verified, matrix operand order checked by reading, not run on hardware
//
#include <hip/hip_runtime.h>
#include <hip/hip_bf16.h>

#define L_SEQ 2048
#define BATCH 2
#define DIM 1024
#define HEADS 16
#define HDIM 64
#define ATTN_SCALE 0.125f   // 64^-0.5

typedef __attribute__((ext_vector_type(16))) __bf16 v16bf;
typedef __attribute__((ext_vector_type(8)))  float  v8f;
typedef __attribute__((ext_vector_type(8)))  unsigned short u16x8;
typedef __attribute__((ext_vector_type(4)))  unsigned int v4u;
typedef __attribute__((ext_vector_type(8)))  int v8i;
typedef __attribute__((ext_vector_type(4)))  int v4i;

union BF16Frag { u16x8 h[2]; v16bf v; };

static __device__ __forceinline__ unsigned short f2bf(float f) {
  unsigned int u = __float_as_uint(f);
  u += 0x7FFFu + ((u >> 16) & 1u);          // round-to-nearest-even
  return (unsigned short)(u >> 16);
}

static __device__ __forceinline__ v8f wmma_bf16(const BF16Frag& a, const BF16Frag& b, v8f c) {
  return __builtin_amdgcn_wmma_f32_16x16x32_bf16(false, a.v, false, b.v, (short)0, c, false, false);
}

// ---- DPP16 butterfly reductions over 16-lane groups (no LDS, no dscnt stalls) ----
// quad_perm(xor1)=0xB1, quad_perm(xor2)=0x4E, row_half_mirror=0x141, row_mirror=0x140
// dpp_ctrl must be a constant-expression at the builtin call -> template parameter.

template<int CTRL>
static __device__ __forceinline__ float dpp_step(float x) {
  int t = __builtin_amdgcn_update_dpp(__float_as_int(x), __float_as_int(x),
                                      CTRL, 0xF, 0xF, true);
  return __int_as_float(t);
}
static __device__ __forceinline__ float red_max16(float x) {
  x = fmaxf(x, dpp_step<0xB1>(x));
  x = fmaxf(x, dpp_step<0x4E>(x));
  x = fmaxf(x, dpp_step<0x141>(x));
  x = fmaxf(x, dpp_step<0x140>(x));
  return x;
}
static __device__ __forceinline__ float red_sum16(float x) {
  x += dpp_step<0xB1>(x);
  x += dpp_step<0x4E>(x);
  x += dpp_step<0x141>(x);
  x += dpp_step<0x140>(x);
  return x;
}

// Flat pointer to LDS byte offset (flat LDS aperture keeps offset in addr[31:0]).
static __device__ __forceinline__ unsigned lds_off(const void* p) {
  return (unsigned)(unsigned long long)p;
}

// ---------------- Tensor Data Mover: 2D tile load (bf16 elements) ----------------
// tensor == tile: dim0 elements contiguous, dim1 lines of stride `stride0` elements.
// LDS receives tile row-major [dim1][dim0].
static __device__ __forceinline__ void tdm_load_2d(const void* gptr, unsigned lds_addr,
                                                   unsigned dim0, unsigned dim1,
                                                   unsigned long long stride0) {
  unsigned long long ga = (unsigned long long)gptr;
  v4u g0;
  g0[0] = 1u;                                              // count=1, user mode
  g0[1] = lds_addr;                                        // lds_addr (bytes)
  g0[2] = (unsigned)(ga & 0xffffffffu);                    // global_addr[31:0]
  g0[3] = (unsigned)((ga >> 32) & 0x1ffffffu) | (2u << 30); // global_addr[56:32] | type=2
  v8i g1;
  g1[0] = (int)(1u << 16);                                 // data_size=1 -> 2 bytes
  g1[1] = (int)((dim0 & 0xffffu) << 16);                   // tensor_dim0[15:0]
  g1[2] = (int)(((dim0 >> 16) & 0xffffu) | ((dim1 & 0xffffu) << 16)); // dim0 hi | dim1 lo
  g1[3] = (int)(((dim1 >> 16) & 0xffffu) | (dim0 << 16));  // dim1 hi | tile_dim0
  g1[4] = (int)(dim1 & 0xffffu);                           // tile_dim1 (tile_dim2=0)
  g1[5] = (int)(stride0 & 0xffffffffu);                    // tensor_dim0_stride lo
  g1[6] = (int)((stride0 >> 32) & 0xffffu);                // stride hi (dim1_stride=0)
  g1[7] = 0;
  v4i z4 = {0, 0, 0, 0};
#if defined(__clang_major__) && __clang_major__ >= 23
  v8i z8 = {0, 0, 0, 0, 0, 0, 0, 0};
  __builtin_amdgcn_tensor_load_to_lds(g0, g1, z4, z4, z8, 0);
#else
  __builtin_amdgcn_tensor_load_to_lds(g0, g1, z4, z4, 0);
#endif
}

// ---------------- fp32 -> bf16 converts ----------------

__global__ void cvt_bf16(const float* __restrict__ src, unsigned short* __restrict__ dst, int n) {
  int i = blockIdx.x * blockDim.x + threadIdx.x;
  if (i < n) dst[i] = f2bf(src[i]);
}

// w [K,N] row-major fp32  ->  wT [N,K] row-major bf16
__global__ void cvt_transpose_bf16(const float* __restrict__ w, unsigned short* __restrict__ wT,
                                   int K, int N) {
  int i = blockIdx.x * blockDim.x + threadIdx.x;
  if (i < K * N) {
    int k = i / N, n = i % N;
    wT[n * K + k] = f2bf(w[i]);
  }
}

// ---------------- bf16 GEMM: C[M,N] = A[M,K] * B[K,N] (B given as Bt[N,K]) ----------------
// Block = 4 waves on a 64x64 tile. A and B tiles (64 x 32k) staged via TDM, double-buffered;
// inner loop touches only LDS + WMMA.
// MODE 1: QKV special (q/k row-major, v scattered to vT). MODE 2: fp32 row-major out.

template<int MODE>
__global__ void __launch_bounds__(128)
gemm_bf16(const unsigned short* __restrict__ A, const unsigned short* __restrict__ Bt,
          unsigned short* __restrict__ Cb, float* __restrict__ Cf,
          unsigned short* __restrict__ vT, int M, int N, int K) {
  __shared__ unsigned short As[2][64 * 32];    // [row][k]
  __shared__ unsigned short Bs[2][64 * 32];    // [n][k]

  const int nblk = N >> 6;
  const int bm   = blockIdx.x / nblk;
  const int bn   = blockIdx.x % nblk;
  const int wave = threadIdx.x >> 5;
  const int lane = threadIdx.x & 31;
  const int lh   = lane & 15;
  const int kbase = (lane >> 4) * 8;
  const int nk    = K >> 5;

  if (wave == 0) {
    tdm_load_2d(A  + bm * 64 * K, lds_off(&As[0][0]), 32, 64, (unsigned long long)K);
    tdm_load_2d(Bt + bn * 64 * K, lds_off(&Bs[0][0]), 32, 64, (unsigned long long)K);
  }

  v8f acc[4] = {};
  for (int ki = 0; ki < nk; ++ki) {
    __syncthreads();                       // prior consumers of buf (ki+1)&1 are done
    if (wave == 0) {
      if (ki + 1 < nk) {
        tdm_load_2d(A  + bm * 64 * K + (ki + 1) * 32, lds_off(&As[(ki + 1) & 1][0]),
                    32, 64, (unsigned long long)K);
        tdm_load_2d(Bt + bn * 64 * K + (ki + 1) * 32, lds_off(&Bs[(ki + 1) & 1][0]),
                    32, 64, (unsigned long long)K);
        __builtin_amdgcn_s_wait_tensorcnt(2);   // pair ki landed, prefetch in flight
      } else {
        __builtin_amdgcn_s_wait_tensorcnt(0);
      }
    }
    __syncthreads();

    const unsigned short* Asb = As[ki & 1];
    const unsigned short* Bsb = Bs[ki & 1];
    BF16Frag a;
    const unsigned short* ap = Asb + (wave * 16 + lh) * 32 + kbase;
    a.h[0] = *(const u16x8*)(ap);
    a.h[1] = *(const u16x8*)(ap + 16);
#pragma unroll
    for (int j = 0; j < 4; ++j) {
      const unsigned short* bp = Bsb + (j * 16 + lh) * 32 + kbase;
      BF16Frag b;
      b.h[0] = *(const u16x8*)(bp);
      b.h[1] = *(const u16x8*)(bp + 16);
      acc[j] = wmma_bf16(a, b, acc[j]);
    }
  }

  // C layout: VGPR i -> row i (lanes 0-15) / i+8 (lanes 16-31), lane&15 -> col
#pragma unroll
  for (int j = 0; j < 4; ++j) {
#pragma unroll
    for (int i = 0; i < 8; ++i) {
      const int r = bm * 64 + wave * 16 + ((lane < 16) ? i : i + 8);
      const int c = bn * 64 + j * 16 + lh;
      if (MODE == 2) {
        Cf[r * N + c] = acc[j][i];
      } else {               // MODE 1: qkv
        if (c < 2 * DIM) {
          Cb[r * N + c] = f2bf(acc[j][i]);     // Q and K: row-major [B*L, 3*DIM]
        } else {                                // V: scatter into vT [B][H][d][L]
          const int cc = c - 2 * DIM;
          const int hh = cc >> 6, dd = cc & 63;
          const int bb = r >> 11, ll = r & (L_SEQ - 1);
          vT[((bb * HEADS + hh) * HDIM + dd) * L_SEQ + ll] = f2bf(acc[j][i]);
        }
      }
    }
  }
}

// ---------------- flash attention ----------------
// qkv: bf16 [B*L, 3*DIM] (V region unused); vT: bf16 [B][H][64][L]; ob: bf16 [B*L, DIM]
// Block = 4 waves = 64 query rows of one (b,h). 32-key chunks via double-buffered TDM.

__global__ void __launch_bounds__(128)
flash_attn(const unsigned short* __restrict__ qkv, const unsigned short* __restrict__ vT,
           unsigned short* __restrict__ ob) {
  const int nqb  = L_SEQ / 64;                 // 32
  const int qb   = blockIdx.x % nqb;
  const int h    = (blockIdx.x / nqb) % HEADS;
  const int b    = blockIdx.x / (nqb * HEADS);
  const int wave = threadIdx.x >> 5;
  const int lane = threadIdx.x & 31;
  const int lh   = lane & 15;
  const int kbase = (lane >> 4) * 8;

  __shared__ unsigned short Ks[2][32 * 64];    // [key][d]
  __shared__ unsigned short Vt[2][64 * 32];    // [d][key]
  __shared__ unsigned short Pl[4][16 * 32];    // per-wave P scratch [row][key]

  const int qrow0 = qb * 64 + wave * 16;

  // Q fragments (A-matrix 16x32, two K-steps covering d=0..63)
  BF16Frag qa[2];
  {
    const int base = (b * L_SEQ + qrow0 + lh) * (3 * DIM) + h * HDIM;
#pragma unroll
    for (int ks = 0; ks < 2; ++ks) {
      const unsigned short* qp = qkv + base + ks * 32 + kbase;
      qa[ks].h[0] = *(const u16x8*)(qp);
      qa[ks].h[1] = *(const u16x8*)(qp + 16);
    }
  }

  v8f o[4] = {};
  float m[8], lsum[8];
#pragma unroll
  for (int i = 0; i < 8; ++i) { m[i] = -__builtin_inff(); lsum[i] = 0.f; }

  const int NK = L_SEQ / 32;                   // 64 key chunks
  const unsigned long long vhead = (unsigned long long)((b * HEADS + h) * HDIM) * L_SEQ;

  if (wave == 0) {
    tdm_load_2d(qkv + (b * L_SEQ) * (3 * DIM) + DIM + h * HDIM, lds_off(&Ks[0][0]),
                64, 32, 3 * DIM);
    tdm_load_2d(vT + vhead, lds_off(&Vt[0][0]), 32, 64, L_SEQ);
  }

  for (int kc = 0; kc < NK; ++kc) {
    __syncthreads();                           // prior consumers of buf (kc+1)&1 done
    if (wave == 0) {
      if (kc + 1 < NK) {
        const int kb2 = (kc + 1) * 32;
        tdm_load_2d(qkv + (b * L_SEQ + kb2) * (3 * DIM) + DIM + h * HDIM,
                    lds_off(&Ks[(kc + 1) & 1][0]), 64, 32, 3 * DIM);
        tdm_load_2d(vT + vhead + kb2, lds_off(&Vt[(kc + 1) & 1][0]), 32, 64, L_SEQ);
        __builtin_amdgcn_s_wait_tensorcnt(2);  // chunk kc landed, next pair in flight
      } else {
        __builtin_amdgcn_s_wait_tensorcnt(0);
      }
    }
    __syncthreads();

    const unsigned short* Ksb = Ks[kc & 1];
    const unsigned short* Vtb = Vt[kc & 1];

    // scores for two 16-key tiles: S = Q * K^T  (2 WMMAs each over d)
    v8f s[2];
#pragma unroll
    for (int t = 0; t < 2; ++t) {
      const unsigned short* kp = Ksb + (t * 16 + lh) * 64;
      BF16Frag bk0, bk1;
      bk0.h[0] = *(const u16x8*)(kp + kbase);
      bk0.h[1] = *(const u16x8*)(kp + kbase + 16);
      bk1.h[0] = *(const u16x8*)(kp + 32 + kbase);
      bk1.h[1] = *(const u16x8*)(kp + 32 + kbase + 16);
      v8f acc = {};
      acc = wmma_bf16(qa[0], bk0, acc);
      acc = wmma_bf16(qa[1], bk1, acc);
      s[t] = acc;
    }

    // online softmax; row r of the 16x32 score slab lives in VGPR i across a 16-lane half
    float scalef[8];
#pragma unroll
    for (int t = 0; t < 2; ++t)
#pragma unroll
      for (int i = 0; i < 8; ++i) s[t][i] *= ATTN_SCALE;

#pragma unroll
    for (int i = 0; i < 8; ++i) {
      const float mx = red_max16(fmaxf(s[0][i], s[1][i]));
      const float mn = fmaxf(m[i], mx);
      const float sc = __expf(m[i] - mn);
      m[i] = mn; scalef[i] = sc;
      const float p0 = __expf(s[0][i] - mn);
      const float p1 = __expf(s[1][i] - mn);
      s[0][i] = p0; s[1][i] = p1;
      lsum[i] = lsum[i] * sc + red_sum16(p0 + p1);
    }
#pragma unroll
    for (int j = 0; j < 4; ++j)
#pragma unroll
      for (int i = 0; i < 8; ++i) o[j][i] *= scalef[i];

    // P: C-layout -> A-layout via per-wave LDS bounce (same-wave LDS is in-order)
    unsigned short* pw = Pl[wave];
#pragma unroll
    for (int t = 0; t < 2; ++t)
#pragma unroll
      for (int i = 0; i < 8; ++i) {
        const int r = (lane < 16) ? i : i + 8;
        pw[r * 32 + t * 16 + lh] = f2bf(s[t][i]);
      }
    BF16Frag pf;
    const unsigned short* pr = pw + lh * 32 + kbase;
    pf.h[0] = *(const u16x8*)(pr);
    pf.h[1] = *(const u16x8*)(pr + 16);

    // O += P * V  (4 accumulators across d)
#pragma unroll
    for (int nt = 0; nt < 4; ++nt) {
      const unsigned short* vp = Vtb + (nt * 16 + lh) * 32 + kbase;
      BF16Frag vf;
      vf.h[0] = *(const u16x8*)(vp);
      vf.h[1] = *(const u16x8*)(vp + 16);
      o[nt] = wmma_bf16(pf, vf, o[nt]);
    }
  }

  // epilogue: normalize, store bf16 O [B*L, DIM]
#pragma unroll
  for (int i = 0; i < 8; ++i) lsum[i] = 1.f / lsum[i];
#pragma unroll
  for (int nt = 0; nt < 4; ++nt)
#pragma unroll
    for (int i = 0; i < 8; ++i) {
      const int r = qrow0 + ((lane < 16) ? i : i + 8);
      const int c = h * HDIM + nt * 16 + lh;
      ob[(b * L_SEQ + r) * DIM + c] = f2bf(o[nt][i] * lsum[i]);
    }
}

// ---------------- launch ----------------

extern "C" void kernel_launch(void* const* d_in, const int* in_sizes, int n_in,
                              void* d_out, int out_size, void* d_ws, size_t ws_size,
                              hipStream_t stream) {
  const float* x     = (const float*)d_in[0];   // [2,2048,1024]
  const float* w_qkv = (const float*)d_in[1];   // [1024,3072]
  const float* w_out = (const float*)d_in[2];   // [1024,1024]
  float* out = (float*)d_out;                   // [2,2048,1024]

  const int M = BATCH * L_SEQ;                  // 4096
  unsigned short* xb    = (unsigned short*)d_ws;           // [4096,1024]
  unsigned short* wqkvT = xb    + M * DIM;                 // [3072,1024]
  unsigned short* woutT = wqkvT + DIM * 3 * DIM;           // [1024,1024]
  unsigned short* qkv   = woutT + DIM * DIM;               // [4096,3072] (V third unused)
  unsigned short* vTb   = qkv   + M * 3 * DIM;             // [2,16,64,2048]
  unsigned short* obuf  = vTb   + M * DIM;                 // [4096,1024]

  cvt_bf16<<<(M * DIM) / 256, 256, 0, stream>>>(x, xb, M * DIM);
  cvt_transpose_bf16<<<(DIM * 3 * DIM) / 256, 256, 0, stream>>>(w_qkv, wqkvT, DIM, 3 * DIM);
  cvt_transpose_bf16<<<(DIM * DIM) / 256, 256, 0, stream>>>(w_out, woutT, DIM, DIM);

  // QKV GEMM: (4096/64)*(3072/64) = 64*48 = 3072 blocks, 64x64 tile each
  gemm_bf16<1><<<3072, 128, 0, stream>>>(xb, wqkvT, qkv, nullptr, vTb, M, 3 * DIM, DIM);

  // attention: B*H*(L/64) = 2*16*32 = 1024 blocks
  flash_attn<<<BATCH * HEADS * (L_SEQ / 64), 128, 0, stream>>>(qkv, vTb, obuf);

  // out-proj: (4096/64)*(1024/64) = 64*16 = 1024 blocks, fp32 output
  gemm_bf16<2><<<1024, 128, 0, stream>>>(obuf, woutT, nullptr, out, nullptr, M, DIM, DIM);
}
